// AttentionONNX_6116033429975
// MI455X (gfx1250) — matrix operand order repaired and checked by reading, not compiled, not run on hardware
//
#include <hip/hip_runtime.h>

// Problem constants (match reference)
#define B_ 2
#define T_ 2048
#define D_ 1024
#define H_ 16
#define HD_ 64
#define SCALE_ 0.125f
#define WINDOW_ 128
#define NEG_INF_ (-1.0e9f)

typedef _Float16 h16 __attribute__((ext_vector_type(16)));
typedef float f32x8 __attribute__((ext_vector_type(8)));
typedef float f32x4 __attribute__((ext_vector_type(4)));

union Frag {
  h16   h;      // 16 halves = 8 VGPRs (WMMA A/B fragment)
  f32x4 q[2];   // two 16-byte chunks for b128 loads
};

static __device__ __forceinline__ f32x8 wmma16(const Frag& a, const Frag& b, f32x8 c) {
  // (neg_a, A, neg_b, B, c_mod, C, reuse_a, reuse_b)
  return __builtin_amdgcn_wmma_f32_16x16x32_f16(false, a.h, false, b.h, (short)0, c, false, false);
}

// --- fragment loaders for the 32x64 GEMM tile -------------------------------
static __device__ __forceinline__ void load_a2(Frag* a, const _Float16* xlo,
                                               const _Float16* xhi, int kc) {
  const _Float16* p0 = xlo + kc;
  a[0].q[0] = *(const f32x4*)(p0);
  a[0].q[1] = *(const f32x4*)(p0 + 16);
  const _Float16* p1 = xhi + kc;
  a[1].q[0] = *(const f32x4*)(p1);
  a[1].q[1] = *(const f32x4*)(p1 + 16);
}

static __device__ __forceinline__ void load_b4(Frag* bf, const _Float16* wrow, int kc) {
  const _Float16* bp = wrow + kc;
#pragma unroll
  for (int nt = 0; nt < 4; ++nt) {
    const _Float16* p = bp + (size_t)nt * 16 * D_;
    bf[nt].q[0] = *(const f32x4*)(p);
    bf[nt].q[1] = *(const f32x4*)(p + 16);
  }
}

static __device__ __forceinline__ void mma8(f32x8 acc[2][4], const Frag* a, const Frag* bf) {
#pragma unroll
  for (int mr = 0; mr < 2; ++mr)
#pragma unroll
    for (int nt = 0; nt < 4; ++nt)
      acc[mr][nt] = wmma16(a[mr], bf[nt], acc[mr][nt]);
}

// ---------------------------------------------------------------------------
// fp32 -> fp16 conversion, 4 elements / thread
// ---------------------------------------------------------------------------
__global__ void cvt_f32_f16(const float* __restrict__ in, _Float16* __restrict__ out, int n4) {
  int i = blockIdx.x * blockDim.x + threadIdx.x;
  if (i < n4) {
    f32x4 v = ((const f32x4*)in)[i];
    union { _Float16 h[4]; unsigned long long u; } p;
    p.h[0] = (_Float16)v[0];
    p.h[1] = (_Float16)v[1];
    p.h[2] = (_Float16)v[2];
    p.h[3] = (_Float16)v[3];
    ((unsigned long long*)out)[i] = p.u;
  }
}

// ---------------------------------------------------------------------------
// Fused QKV projection + RoPE.  One wave = 32 tokens x 64 cols (one head),
// double-buffered fragment pipeline.  grid: (BT/32, H, 3) block: 32.
//   z=0 -> q (rope, store [B*H][T][64])
//   z=1 -> k (rope, store [B*H][T][64])
//   z=2 -> v (store transposed [B*H][64][T])
// ---------------------------------------------------------------------------
__global__ __launch_bounds__(32)
void qkv_rope_kernel(const _Float16* __restrict__ x16,
                     const _Float16* __restrict__ wq16,
                     const _Float16* __restrict__ wk16,
                     const _Float16* __restrict__ wv16,
                     const float* __restrict__ rope_cos,
                     const float* __restrict__ rope_sin,
                     _Float16* __restrict__ q16,
                     _Float16* __restrict__ k16,
                     _Float16* __restrict__ vT16)
{
  const int lane = threadIdx.x;
  const int nl   = lane & 15;
  const int half = lane >> 4;
  const int mt   = blockIdx.x;        // 32-token tile over B*T
  const int hsel = blockIdx.y;        // head index = 64-col strip
  const int z    = blockIdx.z;
  const _Float16* __restrict__ w16 = (z == 0) ? wq16 : (z == 1) ? wk16 : wv16;

  const int row0  = mt * 32;
  const int nbase = hsel * 64;

  const _Float16* xlo  = x16 + (size_t)(row0 + nl) * D_ + 8 * half;
  const _Float16* xhi  = x16 + (size_t)(row0 + 16 + nl) * D_ + 8 * half;
  const _Float16* wrow = w16 + (size_t)(nbase + nl) * D_ + 8 * half;

  f32x8 acc[2][4] = {};               // 32x64 f32 accumulator

  Frag A0[2], A1[2], B0[4], B1[4];
  load_a2(A0, xlo, xhi, 0);
  load_b4(B0, wrow, 0);

  for (int kc = 0; kc < D_; kc += 64) {
    load_a2(A1, xlo, xhi, kc + 32);
    load_b4(B1, wrow, kc + 32);
    __builtin_prefetch(xlo + kc + 128, 0, 3);
    __builtin_prefetch(wrow + kc + 128, 0, 3);
    mma8(acc, A0, B0);
    if (kc + 64 < D_) {
      load_a2(A0, xlo, xhi, kc + 64);
      load_b4(B0, wrow, kc + 64);
    }
    mma8(acc, A1, B1);
  }

  const int b  = row0 / T_;
  const int t0 = row0 % T_;
  const int bh = b * H_ + hsel;

  if (z < 2) {
    _Float16* __restrict__ dst = (z == 0) ? q16 : k16;
#pragma unroll
    for (int mr = 0; mr < 2; ++mr)
#pragma unroll
      for (int nt = 0; nt < 4; ++nt) {
        const int dimc = nt * 16 + nl;
#pragma unroll
        for (int v = 0; v < 8; ++v) {
          const int t = t0 + mr * 16 + v + 8 * half;
          float val   = acc[mr][nt][v];
          float prt   = __shfl_xor(val, 1, 32);       // rope pair partner
          float c     = rope_cos[t * (HD_ / 2) + (dimc >> 1)];
          float s     = rope_sin[t * (HD_ / 2) + (dimc >> 1)];
          float y     = ((dimc & 1) == 0) ? (val * c - prt * s)
                                          : (prt * s + val * c);
          dst[((size_t)bh * T_ + t) * HD_ + dimc] = (_Float16)y;
        }
      }
  } else {
#pragma unroll
    for (int mr = 0; mr < 2; ++mr)
#pragma unroll
      for (int nt = 0; nt < 4; ++nt) {
        const int dimc = nt * 16 + nl;
#pragma unroll
        for (int v = 0; v < 8; ++v) {
          const int t = t0 + mr * 16 + v + 8 * half;
          vT16[((size_t)bh * HD_ + dimc) * T_ + t] = (_Float16)acc[mr][nt][v];
        }
      }
  }
}

// ---------------------------------------------------------------------------
// Sliding-window causal attention with online softmax.
// One wave per (b,h, 16-query tile).  grid: (T/16, B*H) block: 32.
// Keys processed in 5 chunks of 32 covering [16qt-144, 16qt+15].
// ---------------------------------------------------------------------------
__global__ __launch_bounds__(32)
void attn_kernel(const _Float16* __restrict__ q16,
                 const _Float16* __restrict__ k16,
                 const _Float16* __restrict__ vT16,
                 _Float16* __restrict__ o16)
{
  __shared__ __align__(16) _Float16 pbuf[16 * 32];   // P tile staging (C->A relayout)

  const int lane = threadIdx.x;
  const int nl   = lane & 15;
  const int half = lane >> 4;
  const int qt   = blockIdx.x;
  const int bh   = blockIdx.y;
  const int b    = bh / H_;
  const int h    = bh % H_;
  const int i0   = qt * 16;

  // Q A-fragments for K-dim chunks 0..31 and 32..63
  Frag qa0, qa1;
  {
    const _Float16* qp = q16 + ((size_t)bh * T_ + i0 + nl) * HD_ + 8 * half;
    qa0.q[0] = *(const f32x4*)(qp);
    qa0.q[1] = *(const f32x4*)(qp + 16);
    qa1.q[0] = *(const f32x4*)(qp + 32);
    qa1.q[1] = *(const f32x4*)(qp + 48);
  }

  float mrun[8], srun[8];
  f32x8 o[4] = {};
#pragma unroll
  for (int v = 0; v < 8; ++v) { mrun[v] = NEG_INF_; srun[v] = 0.f; }

  for (int it = 0; it < 5; ++it) {
    const int j0 = i0 - 144 + 32 * it;

    // ---- S = scale * Q K^T (two 16-key tiles) ----
    f32x8 st[2] = {};
#pragma unroll
    for (int t = 0; t < 2; ++t) {
      int j  = j0 + t * 16 + nl;
      int jc = j < 0 ? 0 : (j > T_ - 1 ? T_ - 1 : j);
      const _Float16* kp = k16 + ((size_t)bh * T_ + jc) * HD_ + 8 * half;
      Frag kb0, kb1;
      kb0.q[0] = *(const f32x4*)(kp);
      kb0.q[1] = *(const f32x4*)(kp + 16);
      kb1.q[0] = *(const f32x4*)(kp + 32);
      kb1.q[1] = *(const f32x4*)(kp + 48);
      st[t] = wmma16(qa0, kb0, st[t]);
      st[t] = wmma16(qa1, kb1, st[t]);
    }

    // ---- mask + scale ----
#pragma unroll
    for (int t = 0; t < 2; ++t) {
      const int j = j0 + t * 16 + nl;
#pragma unroll
      for (int v = 0; v < 8; ++v) {
        const int i = i0 + v + 8 * half;
        const bool ok = (j >= 0) && (j <= i) && (j >= i - WINDOW_);
        st[t][v] = ok ? st[t][v] * SCALE_ : NEG_INF_;
      }
    }

    // ---- online softmax: row max / exp / row sum (butterfly in half-wave) ----
    float alpha[8];
#pragma unroll
    for (int v = 0; v < 8; ++v) {
      float rm = fmaxf(st[0][v], st[1][v]);
      rm = fmaxf(rm, __shfl_xor(rm, 1, 32));
      rm = fmaxf(rm, __shfl_xor(rm, 2, 32));
      rm = fmaxf(rm, __shfl_xor(rm, 4, 32));
      rm = fmaxf(rm, __shfl_xor(rm, 8, 32));
      const float mn = fmaxf(mrun[v], rm);
      alpha[v] = __expf(mrun[v] - mn);
      mrun[v]  = mn;
      st[0][v] = __expf(st[0][v] - mn);
      st[1][v] = __expf(st[1][v] - mn);
      float rs = st[0][v] + st[1][v];
      rs += __shfl_xor(rs, 1, 32);
      rs += __shfl_xor(rs, 2, 32);
      rs += __shfl_xor(rs, 4, 32);
      rs += __shfl_xor(rs, 8, 32);
      srun[v] = srun[v] * alpha[v] + rs;
    }
#pragma unroll
    for (int dt = 0; dt < 4; ++dt)
#pragma unroll
      for (int v = 0; v < 8; ++v)
        o[dt][v] *= alpha[v];

    // ---- relayout P (C layout) -> A fragment via LDS ----
#pragma unroll
    for (int t = 0; t < 2; ++t)
#pragma unroll
      for (int v = 0; v < 8; ++v)
        pbuf[(v + 8 * half) * 32 + t * 16 + nl] = (_Float16)st[t][v];
    __syncthreads();
    Frag pa;
    {
      const _Float16* pp = pbuf + nl * 32 + 16 * half;
      pa.q[0] = *(const f32x4*)(pp);
      pa.q[1] = *(const f32x4*)(pp + 8);
    }
    __syncthreads();

    // ---- O += P V  (V stored transposed [bh][d][t] -> contiguous B frags) ----
    const int jv  = j0 + 16 * half;                    // always multiple of 16
    const int jvc = jv < 0 ? 0 : (jv > T_ - 16 ? T_ - 16 : jv);
#pragma unroll
    for (int dt = 0; dt < 4; ++dt) {
      const _Float16* vp = vT16 + ((size_t)bh * HD_ + dt * 16 + nl) * T_ + jvc;
      Frag vb;
      vb.q[0] = *(const f32x4*)(vp);
      vb.q[1] = *(const f32x4*)(vp + 8);
      o[dt] = wmma16(pa, vb, o[dt]);
    }
  }

  // ---- normalize + store f16 into [B*T][D] for the output projection ----
#pragma unroll
  for (int dt = 0; dt < 4; ++dt)
#pragma unroll
    for (int v = 0; v < 8; ++v) {
      const int t = i0 + v + 8 * half;
      const float y = o[dt][v] / srun[v];
      o16[((size_t)(b * T_ + t)) * D_ + h * HD_ + dt * 16 + nl] = (_Float16)y;
    }
}

// ---------------------------------------------------------------------------
// Output projection: out = attn @ wo^T  (fp32 result).
// One wave = 32 rows x 64 cols, double-buffered. grid: (BT/32, D/64) block: 32.
// ---------------------------------------------------------------------------
__global__ __launch_bounds__(32)
void proj_kernel(const _Float16* __restrict__ a16,
                 const _Float16* __restrict__ wo16,
                 float* __restrict__ out)
{
  const int lane = threadIdx.x;
  const int nl   = lane & 15;
  const int half = lane >> 4;
  const int mt   = blockIdx.x;
  const int ns   = blockIdx.y;

  const int row0  = mt * 32;
  const int nbase = ns * 64;

  const _Float16* xlo  = a16 + (size_t)(row0 + nl) * D_ + 8 * half;
  const _Float16* xhi  = a16 + (size_t)(row0 + 16 + nl) * D_ + 8 * half;
  const _Float16* wrow = wo16 + (size_t)(nbase + nl) * D_ + 8 * half;

  f32x8 acc[2][4] = {};

  Frag A0[2], A1[2], B0[4], B1[4];
  load_a2(A0, xlo, xhi, 0);
  load_b4(B0, wrow, 0);

  for (int kc = 0; kc < D_; kc += 64) {
    load_a2(A1, xlo, xhi, kc + 32);
    load_b4(B1, wrow, kc + 32);
    __builtin_prefetch(xlo + kc + 128, 0, 3);
    __builtin_prefetch(wrow + kc + 128, 0, 3);
    mma8(acc, A0, B0);
    if (kc + 64 < D_) {
      load_a2(A0, xlo, xhi, kc + 64);
      load_b4(B0, wrow, kc + 64);
    }
    mma8(acc, A1, B1);
  }

#pragma unroll
  for (int mr = 0; mr < 2; ++mr)
#pragma unroll
    for (int nt = 0; nt < 4; ++nt)
#pragma unroll
      for (int v = 0; v < 8; ++v) {
        const int row = row0 + mr * 16 + v + 8 * half;
        out[(size_t)row * D_ + nbase + nt * 16 + nl] = acc[mr][nt][v];
      }
}

// ---------------------------------------------------------------------------
extern "C" void kernel_launch(void* const* d_in, const int* in_sizes, int n_in,
                              void* d_out, int out_size, void* d_ws, size_t ws_size,
                              hipStream_t stream) {
  (void)in_sizes; (void)n_in; (void)out_size; (void)ws_size;

  const float* x        = (const float*)d_in[0];
  const float* rope_cos = (const float*)d_in[1];
  const float* rope_sin = (const float*)d_in[2];
  const float* wq       = (const float*)d_in[3];
  const float* wk       = (const float*)d_in[4];
  const float* wv       = (const float*)d_in[5];
  const float* wo       = (const float*)d_in[6];

  char* ws = (char*)d_ws;
  const size_t MB = 1u << 20;
  _Float16* x16  = (_Float16*)(ws);             //  8 MB : x in f16
  _Float16* wq16 = (_Float16*)(ws +  8 * MB);   //  2 MB
  _Float16* wk16 = (_Float16*)(ws + 10 * MB);   //  2 MB
  _Float16* wv16 = (_Float16*)(ws + 12 * MB);   //  2 MB
  _Float16* wo16 = (_Float16*)(ws + 14 * MB);   //  2 MB
  _Float16* q16  = (_Float16*)(ws + 16 * MB);   //  8 MB : [B*H][T][64] (rope'd)
  _Float16* k16  = (_Float16*)(ws + 24 * MB);   //  8 MB : [B*H][T][64] (rope'd)
  _Float16* vT16 = (_Float16*)(ws + 32 * MB);   //  8 MB : [B*H][64][T]
  _Float16* a16  = (_Float16*)(ws + 40 * MB);   //  8 MB : attention out [B*T][D]

  const int nx4 = (B_ * T_ * D_) / 4;
  const int nw4 = (D_ * D_) / 4;
  cvt_f32_f16<<<(nx4 + 255) / 256, 256, 0, stream>>>(x,  x16,  nx4);
  cvt_f32_f16<<<(nw4 + 255) / 256, 256, 0, stream>>>(wq, wq16, nw4);
  cvt_f32_f16<<<(nw4 + 255) / 256, 256, 0, stream>>>(wk, wk16, nw4);
  cvt_f32_f16<<<(nw4 + 255) / 256, 256, 0, stream>>>(wv, wv16, nw4);
  cvt_f32_f16<<<(nw4 + 255) / 256, 256, 0, stream>>>(wo, wo16, nw4);

  qkv_rope_kernel<<<dim3((B_ * T_) / 32, H_, 3), 32, 0, stream>>>(
      x16, wq16, wk16, wv16, rope_cos, rope_sin, q16, k16, vT16);

  attn_kernel<<<dim3(T_ / 16, B_ * H_), 32, 0, stream>>>(q16, k16, vT16, a16);

  proj_kernel<<<dim3((B_ * T_) / 32, D_ / 64), 32, 0, stream>>>(
      a16, wo16, (float*)d_out);
}
